// MultiHeadAttention_35931696398602
// MI455X (gfx1250) — matrix operand order repaired and checked by reading, not compiled, not run on hardware
//
#include <hip/hip_runtime.h>
#include <hip/hip_bf16.h>

typedef unsigned short ushort_t;
typedef __attribute__((ext_vector_type(16))) __bf16 v16bf;
typedef __attribute__((ext_vector_type(8)))  __bf16 v8bf;
typedef __attribute__((ext_vector_type(8)))  float  v8f;

#define TT  2048
#define DD  2048
#define HH  16
#define HDD 128
#define N3  6144

#if __has_builtin(__builtin_amdgcn_global_load_async_to_lds_b128)
#define HAS_ASYNC_LDS 1
#else
#define HAS_ASYNC_LDS 0
#endif

// pointer types for the async builtin (from hipcc diagnostic: param is
// "int __vector(4) __device__ *", i.e. vector_size(16) int in AS1)
typedef int v4i_vs __attribute__((vector_size(16)));
typedef __attribute__((address_space(1))) v4i_vs* gptr_v4i;
typedef __attribute__((address_space(3))) v4i_vs* lptr_v4i;

__device__ __forceinline__ float bf2f(ushort_t u) {
  return __uint_as_float(((unsigned)u) << 16);
}
__device__ __forceinline__ ushort_t f2bf(float f) {
  unsigned u = __float_as_uint(f);
  u += 0x7FFFu + ((u >> 16) & 1u);   // round-to-nearest-even
  return (ushort_t)(u >> 16);
}

// ---- CDNA5 async global->LDS copy (16B per lane), ASYNCcnt tracked ----
__device__ __forceinline__ void cp_async_b128(void* lds, const void* gmem) {
#if HAS_ASYNC_LDS
  __builtin_amdgcn_global_load_async_to_lds_b128(
      (gptr_v4i)gmem, (lptr_v4i)lds, 0, 0);
#else
  *(uint4*)lds = *(const uint4*)gmem;
#endif
}
__device__ __forceinline__ void wait_async0() {
#if __has_builtin(__builtin_amdgcn_s_wait_asynccnt)
  __builtin_amdgcn_s_wait_asynccnt(0);
#else
  asm volatile("s_wait_asynccnt 0x0" ::: "memory");
#endif
}

__device__ __forceinline__ v8f wmma_bf16(v16bf a, v16bf b, v8f c) {
  return __builtin_amdgcn_wmma_f32_16x16x32_bf16(
      false, a, false, b, (short)0, c, false, false);
}

__device__ __forceinline__ v16bf load_frag_2x8(const ushort_t* p0, const ushort_t* p1) {
  v8bf lo = *(const v8bf*)p0;
  v8bf hi = *(const v8bf*)p1;
  v16bf r;
#pragma unroll
  for (int i = 0; i < 8; ++i) { r[i] = lo[i]; r[i + 8] = hi[i]; }
  return r;
}
// A fragment (16x32 bf16): lane%16 = row M; lane holds K [g*8,g*8+8) and [16+g*8,+8)
__device__ __forceinline__ v16bf load_a_frag(const ushort_t* base, int lda, int lane) {
  int m = lane & 15, g = lane >> 4;
  const ushort_t* p = base + m * lda + g * 8;
  return load_frag_2x8(p, p + 16);
}
// B fragment (32x16 bf16): lane%16 = col N; lane holds K [g*8,g*8+8) and [16+g*8,+8); tile stored [n][k]
__device__ __forceinline__ v16bf load_b_frag(const ushort_t* base, int ldb, int lane) {
  int n = lane & 15, g = lane >> 4;
  const ushort_t* p = base + n * ldb + g * 8;
  return load_frag_2x8(p, p + 16);
}

// ---------------------------------------------------------------- convert (straight)
__global__ void k_f32_to_bf16(const float* __restrict__ src,
                              ushort_t* __restrict__ dst, int n) {
  int i = (blockIdx.x * blockDim.x + threadIdx.x) * 4;
  if (i < n) {
    float4 v = *(const float4*)(src + i);
    union { ushort_t u[4]; uint2 v2; } o;
    o.u[0] = f2bf(v.x); o.u[1] = f2bf(v.y);
    o.u[2] = f2bf(v.z); o.u[3] = f2bf(v.w);
    *(uint2*)(dst + i) = o.v2;
  }
}

// ---------------------------------------------------------------- convert + transpose
// out[n][k] (bf16, ld=K) = in[k][n] (f32, ld=N); 64x64 tiles, coalesced both sides
__global__ void __launch_bounds__(256)
k_conv_transpose(const float* __restrict__ in, ushort_t* __restrict__ out,
                 int K, int N) {
  __shared__ float tile[64][65];
  const int k0 = blockIdx.y * 64, n0 = blockIdx.x * 64;
  const int tr = threadIdx.x >> 4;          // 0..15
  const int tc = (threadIdx.x & 15) * 4;    // 0..60
#pragma unroll
  for (int i = 0; i < 4; ++i) {
    int k = tr + i * 16;
    float4 v = *(const float4*)(in + (size_t)(k0 + k) * N + n0 + tc);
    tile[k][tc + 0] = v.x; tile[k][tc + 1] = v.y;
    tile[k][tc + 2] = v.z; tile[k][tc + 3] = v.w;
  }
  __syncthreads();
  const int nr = threadIdx.x >> 2;          // 0..63
  const int kc = (threadIdx.x & 3) * 16;    // 0..48
#pragma unroll
  for (int half = 0; half < 2; ++half) {
    union { ushort_t u[8]; uint4 v; } o;
#pragma unroll
    for (int i = 0; i < 8; ++i) o.u[i] = f2bf(tile[kc + half * 8 + i][nr]);
    *(uint4*)(out + (size_t)(n0 + nr) * K + k0 + kc + half * 8) = o.v;
  }
}

// ---------------------------------------------------------------- RoPE (in-place on q/k)
__global__ void k_rope(ushort_t* __restrict__ qkvb,
                       const float* __restrict__ sin_t,
                       const float* __restrict__ cos_t) {
  int idx = blockIdx.x * blockDim.x + threadIdx.x;  // over T*H*64
  int off = blockIdx.y ? DD : 0;                    // q or k section
  int i = idx & 63;
  int h = (idx >> 6) & (HH - 1);
  int t = idx >> 10;
  ushort_t* p = qkvb + (size_t)t * N3 + off + h * HDD;
  float x1 = bf2f(p[i]);
  float x2 = bf2f(p[i + 64]);
  const float* sr = sin_t + t * HDD;
  const float* cr = cos_t + t * HDD;
  float o1 = x1 * cr[i]      - x2 * sr[i];
  float o2 = x2 * cr[i + 64] + x1 * sr[i + 64];
  p[i]      = f2bf(o1);
  p[i + 64] = f2bf(o2);
}

// ---------------------------------------------------------------- bf16 GEMM
// C[M,N] = A[M,K] (row-major) * Bt[N,K] (row-major = B transposed)
// block 128x128, 8 waves (2M x 4N), wave = 4x2 frags, BLK_K=64,
// async double-buffered LDS staging.
template <bool OUT_F32>
__global__ void __launch_bounds__(256)
k_gemm_bf16(const ushort_t* __restrict__ A, const ushort_t* __restrict__ Bt,
            float* __restrict__ Cf, ushort_t* __restrict__ Cb,
            int M, int N, int K) {
  __shared__ __align__(16) ushort_t As[2][128][72];   // [buf][m][k]
  __shared__ __align__(16) ushort_t Bs[2][128][72];   // [buf][n][k]

  const int tid = threadIdx.x;
  const int lane = tid & 31, wave = tid >> 5;
  const int wm = wave >> 2, wn = wave & 3;
  const int rowBase = blockIdx.y * 128, colBase = blockIdx.x * 128;

  // staging map: each thread copies 32 contiguous bf16 (4 x b128) per tile side
  const int sr = tid >> 1, sc = (tid & 1) * 32;

  const ushort_t* gA = A  + (size_t)(rowBase + sr) * K + sc;
  const ushort_t* gB = Bt + (size_t)(colBase + sr) * K + sc;

  v8f acc[4][2];
#pragma unroll
  for (int mi = 0; mi < 4; ++mi)
#pragma unroll
    for (int ni = 0; ni < 2; ++ni)
#pragma unroll
      for (int j = 0; j < 8; ++j) acc[mi][ni][j] = 0.f;

  auto stage = [&](int buf, int kt) {
    const ushort_t* pa = gA + kt * 64;
    const ushort_t* pb = gB + kt * 64;
#pragma unroll
    for (int i = 0; i < 4; ++i) {
      cp_async_b128(&As[buf][sr][sc + i * 8], pa + i * 8);
      cp_async_b128(&Bs[buf][sr][sc + i * 8], pb + i * 8);
    }
  };

  stage(0, 0);
  const int ksteps = K >> 6;
  for (int kt = 0; kt < ksteps; ++kt) {
    wait_async0();
    __syncthreads();
    if (kt + 1 < ksteps) stage((kt + 1) & 1, kt + 1);
    const int buf = kt & 1;
#pragma unroll
    for (int kk = 0; kk < 2; ++kk) {
      v16bf bfrag[2];
#pragma unroll
      for (int ni = 0; ni < 2; ++ni)
        bfrag[ni] = load_b_frag(&Bs[buf][wn * 32 + ni * 16][kk * 32], 72, lane);
#pragma unroll
      for (int mi = 0; mi < 4; ++mi) {
        v16bf afrag = load_a_frag(&As[buf][wm * 64 + mi * 16][kk * 32], 72, lane);
#pragma unroll
        for (int ni = 0; ni < 2; ++ni)
          acc[mi][ni] = wmma_bf16(afrag, bfrag[ni], acc[mi][ni]);
      }
    }
  }

  // epilogue: C layout -> row = j + 8*(lane/16), col = lane%16
  const int g = lane >> 4, nloc = lane & 15;
#pragma unroll
  for (int mi = 0; mi < 4; ++mi)
#pragma unroll
    for (int ni = 0; ni < 2; ++ni)
#pragma unroll
      for (int j = 0; j < 8; ++j) {
        int row = rowBase + wm * 64 + mi * 16 + j + 8 * g;
        int col = colBase + wn * 32 + ni * 16 + nloc;
        if constexpr (OUT_F32)
          Cf[(size_t)row * N + col] = acc[mi][ni][j];
        else
          Cb[(size_t)row * N + col] = f2bf(acc[mi][ni][j]);
      }
}

// ---------------------------------------------------------------- flash attention
// grid (T/128, H); block = 256 (8 waves). Wave w owns q-rows w*16..w*16+15.
__global__ void __launch_bounds__(256, 1)
k_attn(const ushort_t* __restrict__ qkvb, ushort_t* __restrict__ ob) {
  __shared__ __align__(16) ushort_t Qs[128][HDD + 8];   // [row][d]
  __shared__ __align__(16) ushort_t Ks[128][HDD + 8];   // [key][d]
  __shared__ __align__(16) ushort_t Vs[HDD][128 + 8];   // [d][key] (transposed)
  __shared__ __align__(16) ushort_t Ps[8][16][128 + 8]; // per-wave P staging

  const int tid = threadIdx.x, lane = tid & 31, wave = tid >> 5;
  const int qi = blockIdx.x, h = blockIdx.y;
  const int g = lane >> 4, nl = lane & 15;
  const float scale = 0.08838834764831845f;  // 1/sqrt(128)

  // async-load Q tile (RoPE already applied in-place)
  for (int u = tid; u < 128 * 16; u += 256) {
    int r = u >> 4, dc = (u & 15) * 8;
    cp_async_b128(&Qs[r][dc],
                  qkvb + (size_t)(qi * 128 + r) * N3 + h * HDD + dc);
  }

  float mrow[8], lrow[8];
  v8f o[8];
#pragma unroll
  for (int j = 0; j < 8; ++j) { mrow[j] = -1e30f; lrow[j] = 0.f; }
#pragma unroll
  for (int di = 0; di < 8; ++di)
#pragma unroll
    for (int j = 0; j < 8; ++j) o[di][j] = 0.f;

  for (int kj = 0; kj <= qi; ++kj) {
    __syncthreads();
    // K tile: async straight copy; V tile: transpose scatter via registers
    for (int u = tid; u < 128 * 16; u += 256) {
      int r = u >> 4, dc = (u & 15) * 8;
      const size_t rowoff = (size_t)(kj * 128 + r) * N3 + h * HDD + dc;
      cp_async_b128(&Ks[r][dc], qkvb + rowoff + DD);
      union { uint4 v; ushort_t us[8]; } tv;
      tv.v = *(const uint4*)(qkvb + rowoff + 2 * DD);
#pragma unroll
      for (int i = 0; i < 8; ++i) Vs[dc + i][r] = tv.us[i];
    }
    if (kj < qi)  // hint next K tile toward the caches
      __builtin_prefetch(qkvb + (size_t)((kj + 1) * 128 + (tid >> 1)) * N3 + DD + h * HDD, 0, 0);
    wait_async0();
    __syncthreads();

    // S = Q * K^T over this 128-key tile: 8 N-frags x 4 K-chunks
    v8f s[8];
#pragma unroll
    for (int ni = 0; ni < 8; ++ni)
#pragma unroll
      for (int j = 0; j < 8; ++j) s[ni][j] = 0.f;
#pragma unroll
    for (int kk = 0; kk < 4; ++kk) {
      v16bf afrag = load_a_frag(&Qs[wave * 16][kk * 32], HDD + 8, lane);
#pragma unroll
      for (int ni = 0; ni < 8; ++ni) {
        v16bf bfrag = load_b_frag(&Ks[ni * 16][kk * 32], HDD + 8, lane);
        s[ni] = wmma_bf16(afrag, bfrag, s[ni]);
      }
    }

    // scale + causal mask on the diagonal tile
    const bool diag = (kj == qi);
#pragma unroll
    for (int ni = 0; ni < 8; ++ni)
#pragma unroll
      for (int j = 0; j < 8; ++j) {
        float v = s[ni][j] * scale;
        if (diag) {
          int rloc = wave * 16 + j + 8 * g;
          int cloc = ni * 16 + nl;
          if (cloc > rloc) v = -1e30f;
        }
        s[ni][j] = v;
      }

    // online softmax: row max (across frags, then across 16 lanes)
    float mnew[8];
#pragma unroll
    for (int j = 0; j < 8; ++j) {
      float mx = mrow[j];
#pragma unroll
      for (int ni = 0; ni < 8; ++ni) mx = fmaxf(mx, s[ni][j]);
#pragma unroll
      for (int off = 1; off < 16; off <<= 1)
        mx = fmaxf(mx, __shfl_xor(mx, off, 32));
      mnew[j] = mx;
    }

    // P = exp(S - m), row sums, stage P (bf16) for A-fragment reuse
    float rsum[8];
#pragma unroll
    for (int j = 0; j < 8; ++j) rsum[j] = 0.f;
#pragma unroll
    for (int ni = 0; ni < 8; ++ni)
#pragma unroll
      for (int j = 0; j < 8; ++j) {
        float p = __expf(s[ni][j] - mnew[j]);
        rsum[j] += p;
        Ps[wave][j + 8 * g][ni * 16 + nl] = f2bf(p);
      }
#pragma unroll
    for (int j = 0; j < 8; ++j) {
#pragma unroll
      for (int off = 1; off < 16; off <<= 1)
        rsum[j] += __shfl_xor(rsum[j], off, 32);
      float alpha = __expf(mrow[j] - mnew[j]);
      lrow[j] = lrow[j] * alpha + rsum[j];
      mrow[j] = mnew[j];
#pragma unroll
      for (int di = 0; di < 8; ++di) o[di][j] *= alpha;
    }

    // O += P * V
#pragma unroll
    for (int kk = 0; kk < 4; ++kk) {
      v16bf afrag = load_a_frag(&Ps[wave][0][kk * 32], 128 + 8, lane);
#pragma unroll
      for (int di = 0; di < 8; ++di) {
        v16bf bfrag = load_b_frag(&Vs[di * 16][kk * 32], 128 + 8, lane);
        o[di] = wmma_bf16(afrag, bfrag, o[di]);
      }
    }
  }

  // normalize and store bf16 output [T][H*HD]
#pragma unroll
  for (int j = 0; j < 8; ++j) {
    float inv = 1.0f / lrow[j];
    int t = qi * 128 + wave * 16 + j + 8 * g;
#pragma unroll
    for (int di = 0; di < 8; ++di) {
      int d = di * 16 + nl;
      ob[(size_t)t * DD + h * HDD + d] = f2bf(o[di][j] * inv);
    }
  }
}

// ---------------------------------------------------------------- launcher
extern "C" void kernel_launch(void* const* d_in, const int* in_sizes, int n_in,
                              void* d_out, int out_size, void* d_ws, size_t ws_size,
                              hipStream_t stream) {
  const float* x    = (const float*)d_in[0];
  const float* sn   = (const float*)d_in[1];
  const float* cs   = (const float*)d_in[2];
  const float* wqkv = (const float*)d_in[3];
  const float* wout = (const float*)d_in[4];
  float* out = (float*)d_out;

  char* ws = (char*)d_ws;
  ushort_t* xb      = (ushort_t*)ws; ws += (size_t)TT * DD * 2;
  ushort_t* wqkvbT  = (ushort_t*)ws; ws += (size_t)DD * N3 * 2;  // [N3][D]
  ushort_t* woutbT  = (ushort_t*)ws; ws += (size_t)DD * DD * 2;  // [D][D] transposed
  ushort_t* qkvb    = (ushort_t*)ws; ws += (size_t)TT * N3 * 2;
  ushort_t* obuf    = (ushort_t*)ws; ws += (size_t)TT * DD * 2;

  // 1) fp32 -> bf16; weights transposed once so GEMM tiles stage as straight copies
  k_f32_to_bf16<<<(TT * DD / 4) / 256, 256, 0, stream>>>(x, xb, TT * DD);
  dim3 gt1(N3 / 64, DD / 64);
  k_conv_transpose<<<gt1, 256, 0, stream>>>(wqkv, wqkvbT, DD, N3);
  dim3 gt2(DD / 64, DD / 64);
  k_conv_transpose<<<gt2, 256, 0, stream>>>(wout, woutbT, DD, DD);

  // 2) qkv = x @ w_qkv  (2048 x 6144 x 2048), bf16 out
  dim3 g1(N3 / 128, TT / 128);
  k_gemm_bf16<false><<<g1, 256, 0, stream>>>(xb, wqkvbT, nullptr, qkvb, TT, N3, DD);

  // 3) RoPE in-place on q and k sections
  dim3 g2((TT * HH * 64) / 256, 2);
  k_rope<<<g2, 256, 0, stream>>>(qkvb, sn, cs);

  // 4) flash attention per (q-tile, head)
  dim3 g3(TT / 128, HH);
  k_attn<<<g3, 256, 0, stream>>>(qkvb, obuf);

  // 5) out = attn_out @ w_out  (2048 x 2048 x 2048), f32 out
  dim3 g4(DD / 128, TT / 128);
  k_gemm_bf16<true><<<g4, 256, 0, stream>>>(obuf, woutbT, out, nullptr, TT, DD, DD);
}
